// SAGE_module_15908558864473
// MI455X (gfx1250) — matrix operand order repaired and checked by reading, not compiled, not run on hardware
//
#include <hip/hip_runtime.h>
#include <hip/hip_bf16.h>

// ---------------- sizes ----------------
#define N_NODES   100000
#define N_EDGES   1600000
#define DIM       64
#define N_GRAPHS  512
#define OUT_DIM   192
#define BN_EPS    1e-5f

typedef __attribute__((ext_vector_type(16))) _Float16 v16h;
typedef __attribute__((ext_vector_type(8)))  float    v8f;

// ---------------- utility kernels ----------------
__global__ __launch_bounds__(256)
void zero_f32_kernel(float* __restrict__ p, int n) {
    int i = blockIdx.x * 256 + threadIdx.x;
    if (i < n) p[i] = 0.0f;
}

__global__ __launch_bounds__(256)
void cvt_f16_kernel(_Float16* __restrict__ dst, const float* __restrict__ src, int n) {
    int i = blockIdx.x * 256 + threadIdx.x;
    if (i < n) dst[i] = (_Float16)src[i];
}

// degree via edge-parallel atomics
__global__ __launch_bounds__(256)
void deg_kernel(const int* __restrict__ dst, float* __restrict__ deg, int n_edges) {
    int e = blockIdx.x * 256 + threadIdx.x;
    if (e < n_edges) atomicAdd(&deg[dst[e]], 1.0f);
}

// nodes-per-graph counts
__global__ __launch_bounds__(256)
void cnt_kernel(const int* __restrict__ batch, float* __restrict__ cnt, int n_nodes) {
    int i = blockIdx.x * 256 + threadIdx.x;
    if (i < n_nodes) atomicAdd(&cnt[batch[i]], 1.0f);
}

__global__ __launch_bounds__(256)
void invdeg_kernel(float* __restrict__ deg, int n_nodes) {
    int i = blockIdx.x * 256 + threadIdx.x;
    if (i < n_nodes) deg[i] = 1.0f / fmaxf(deg[i], 1.0f);
}

// ---------------- neighbour scatter-add (L2-resident atomics) ----------------
// 16 threads per edge, float4 per thread (256 B per edge row).
__global__ __launch_bounds__(256)
void scatter_kernel(const float* __restrict__ x,
                    const int* __restrict__ src,
                    const int* __restrict__ dst,
                    float* __restrict__ agg, int n_edges) {
    int t = blockIdx.x * 256 + threadIdx.x;
    int e = t >> 4;
    if (e >= n_edges) return;
    int q = (t & 15) << 2;                                  // float offset 0..60
    const float4 v = *(const float4*)(x + (size_t)src[e] * DIM + q);
    float* d = agg + (size_t)dst[e] * DIM + q;
    atomicAdd(d + 0, v.x);
    atomicAdd(d + 1, v.y);
    atomicAdd(d + 2, v.z);
    atomicAdd(d + 3, v.w);
}

// ---------------- fused SAGE GEMM via WMMA ----------------
// h[m,n] = sum_k (agg[m,k]*invdeg[m]) * Wl[n,k] + sum_k x[m,k] * Wr[n,k] + bl[n]
// block = 4 waves; each wave owns one 16-channel output tile of a 16-node tile.
__global__ __launch_bounds__(128)
void sage_gemm_kernel(const float* __restrict__ agg,
                      const float* __restrict__ invdeg,
                      const float* __restrict__ xin,
                      const _Float16* __restrict__ wl16,   // [64][64] row-major (out,in)
                      const _Float16* __restrict__ wr16,
                      const float* __restrict__ bl,
                      float* __restrict__ h) {
    const int node0 = blockIdx.x * 16;                     // 6250 tiles, exact
    const int lane  = threadIdx.x & 31;
    const int ntile = threadIdx.x >> 5;                    // 0..3
    const int m     = lane & 15;
    const int hs    = lane >> 4;                           // half-select

    const float idg = invdeg[node0 + m];
    v8f c = {};

    // --- neighbour-mean path: A = agg * invdeg, B = Wl^T ---
    #pragma unroll
    for (int kb = 0; kb < 2; ++kb) {
        const float* ar = agg + (size_t)(node0 + m) * DIM + kb * 32 + hs * 16;
        v16h a;
        #pragma unroll
        for (int e = 0; e < 8; ++e) a[e]     = (_Float16)(ar[e]      * idg);
        #pragma unroll
        for (int e = 0; e < 8; ++e) a[e + 8] = (_Float16)(ar[e + 8] * idg);
        v16h b = *(const v16h*)(wl16 + (size_t)(ntile * 16 + m) * DIM + kb * 32 + hs * 16);
        c = __builtin_amdgcn_wmma_f32_16x16x32_f16(false, a, false, b, (short)0, c, false, false);
    }
    // --- self path: A = x, B = Wr^T ---
    #pragma unroll
    for (int kb = 0; kb < 2; ++kb) {
        const float* xr = xin + (size_t)(node0 + m) * DIM + kb * 32 + hs * 16;
        v16h a;
        #pragma unroll
        for (int e = 0; e < 8; ++e) a[e]     = (_Float16)xr[e];
        #pragma unroll
        for (int e = 0; e < 8; ++e) a[e + 8] = (_Float16)xr[e + 8];
        v16h b = *(const v16h*)(wr16 + (size_t)(ntile * 16 + m) * DIM + kb * 32 + hs * 16);
        c = __builtin_amdgcn_wmma_f32_16x16x32_f16(false, a, false, b, (short)0, c, false, false);
    }

    const int   n    = ntile * 16 + m;
    const float bias = bl[n];
    #pragma unroll
    for (int r = 0; r < 8; ++r) {
        const int mm = r + hs * 8;
        h[(size_t)(node0 + mm) * DIM + n] = c[r] + bias;
    }
}

// ---------------- batchnorm statistics ----------------
__global__ __launch_bounds__(256)
void bn_stats_kernel(const float* __restrict__ h, float* __restrict__ stats, int n_nodes) {
    __shared__ float s_sum[256];
    __shared__ float s_sq[256];
    const int c   = threadIdx.x & 63;
    const int grp = threadIdx.x >> 6;                      // 0..3
    float sum = 0.0f, sq = 0.0f;
    for (int i = blockIdx.x * 4 + grp; i < n_nodes; i += gridDim.x * 4) {
        float v = h[(size_t)i * DIM + c];
        sum += v; sq += v * v;
    }
    s_sum[threadIdx.x] = sum;
    s_sq[threadIdx.x]  = sq;
    __syncthreads();
    if (grp == 0) {
        sum = s_sum[c] + s_sum[c + 64] + s_sum[c + 128] + s_sum[c + 192];
        sq  = s_sq[c]  + s_sq[c + 64]  + s_sq[c + 128]  + s_sq[c + 192];
        atomicAdd(&stats[c],      sum);
        atomicAdd(&stats[64 + c], sq);
    }
}

__global__ __launch_bounds__(64)
void bn_finalize_kernel(const float* __restrict__ stats,
                        const float* __restrict__ g,
                        const float* __restrict__ b,
                        float* __restrict__ ss) {
    int c = threadIdx.x;
    float mu  = stats[c] / (float)N_NODES;
    float var = stats[64 + c] / (float)N_NODES - mu * mu;
    float sc  = g[c] * rsqrtf(var + BN_EPS);
    ss[c]      = sc;
    ss[64 + c] = b[c] - mu * sc;
}

// fused BN + ReLU + write next-layer input + pooled-sum atomics
__global__ __launch_bounds__(256)
void bn_apply_pool_kernel(const float* __restrict__ h,
                          const float* __restrict__ ss,
                          float* __restrict__ xout,
                          const int* __restrict__ batch,
                          float* __restrict__ pool, int col_off) {
    int idx = blockIdx.x * 256 + threadIdx.x;
    if (idx >= N_NODES * DIM) return;
    int i = idx >> 6, c = idx & 63;
    float v = h[idx] * ss[c] + ss[64 + c];
    v = v > 0.0f ? v : 0.0f;
    xout[idx] = v;
    atomicAdd(&pool[(size_t)batch[i] * OUT_DIM + col_off + c], v);
}

__global__ __launch_bounds__(256)
void pool_final_kernel(const float* __restrict__ pool,
                       const float* __restrict__ cnt,
                       float* __restrict__ out) {
    int idx = blockIdx.x * 256 + threadIdx.x;
    if (idx >= N_GRAPHS * OUT_DIM) return;
    out[idx] = pool[idx] / fmaxf(cnt[idx / OUT_DIM], 1.0f);
}

// ---------------- launch ----------------
static inline size_t align256(size_t x) { return (x + 255) & ~(size_t)255; }

extern "C" void kernel_launch(void* const* d_in, const int* in_sizes, int n_in,
                              void* d_out, int out_size, void* d_ws, size_t ws_size,
                              hipStream_t stream) {
    const float* x     = (const float*)d_in[0];
    const int*   src   = (const int*)d_in[1];               // edge_index row 0
    const int*   dst   = (const int*)d_in[1] + N_EDGES;     // edge_index row 1
    const int*   batch = (const int*)d_in[2];
    const float* Wl[3] = { (const float*)d_in[3],  (const float*)d_in[8],  (const float*)d_in[13] };
    const float* bl[3] = { (const float*)d_in[4],  (const float*)d_in[9],  (const float*)d_in[14] };
    const float* Wr[3] = { (const float*)d_in[5],  (const float*)d_in[10], (const float*)d_in[15] };
    const float* gm[3] = { (const float*)d_in[6],  (const float*)d_in[11], (const float*)d_in[16] };
    const float* bt[3] = { (const float*)d_in[7],  (const float*)d_in[12], (const float*)d_in[17] };
    float* out = (float*)d_out;

    // workspace layout
    char* ws = (char*)d_ws;
    size_t off = 0;
    const size_t featB = (size_t)N_NODES * DIM * sizeof(float);   // 25.6 MB
    float* agg   = (float*)(ws + off); off = align256(off + featB);
    float* hbuf  = (float*)(ws + off); off = align256(off + featB);
    float* fA    = (float*)(ws + off); off = align256(off + featB);
    float* fB    = (float*)(ws + off); off = align256(off + featB);
    float* deg   = (float*)(ws + off); off = align256(off + (size_t)N_NODES * sizeof(float));
    float* pool  = (float*)(ws + off); off = align256(off + (size_t)N_GRAPHS * OUT_DIM * sizeof(float));
    float* cnt   = (float*)(ws + off); off = align256(off + (size_t)N_GRAPHS * sizeof(float));
    float* stats = (float*)(ws + off); off = align256(off + 128 * sizeof(float));
    float* ss    = (float*)(ws + off); off = align256(off + 128 * sizeof(float));
    _Float16* w16[6];
    for (int i = 0; i < 6; ++i) { w16[i] = (_Float16*)(ws + off); off = align256(off + DIM * DIM * sizeof(_Float16)); }

    const int NB_N    = (N_NODES + 255) / 256;
    const int NB_E    = (N_EDGES + 255) / 256;
    const int NB_SCAT = (N_EDGES * 16 + 255) / 256;
    const int NB_FEAT = (N_NODES * DIM + 255) / 256;
    const int NB_POOL = (N_GRAPHS * OUT_DIM + 255) / 256;

    // per-launch re-initialisation of all accumulators (graph-replay safe)
    zero_f32_kernel<<<NB_N, 256, 0, stream>>>(deg, N_NODES);
    zero_f32_kernel<<<(N_GRAPHS + 255) / 256, 256, 0, stream>>>(cnt, N_GRAPHS);
    zero_f32_kernel<<<NB_POOL, 256, 0, stream>>>(pool, N_GRAPHS * OUT_DIM);

    deg_kernel<<<NB_E, 256, 0, stream>>>(dst, deg, N_EDGES);
    cnt_kernel<<<NB_N, 256, 0, stream>>>(batch, cnt, N_NODES);
    invdeg_kernel<<<NB_N, 256, 0, stream>>>(deg, N_NODES);

    // weights -> f16 (B-fragment is then one contiguous 32 B load per lane)
    for (int l = 0; l < 3; ++l) {
        cvt_f16_kernel<<<(DIM * DIM + 255) / 256, 256, 0, stream>>>(w16[2 * l],     Wl[l], DIM * DIM);
        cvt_f16_kernel<<<(DIM * DIM + 255) / 256, 256, 0, stream>>>(w16[2 * l + 1], Wr[l], DIM * DIM);
    }

    const float* xin[3] = { x, fA, fB };
    float*       xout[3] = { fA, fB, fA };

    for (int l = 0; l < 3; ++l) {
        zero_f32_kernel<<<NB_FEAT, 256, 0, stream>>>(agg, N_NODES * DIM);
        scatter_kernel<<<NB_SCAT, 256, 0, stream>>>(xin[l], src, dst, agg, N_EDGES);
        sage_gemm_kernel<<<N_NODES / 16, 128, 0, stream>>>(agg, deg, xin[l],
                                                           w16[2 * l], w16[2 * l + 1],
                                                           bl[l], hbuf);
        zero_f32_kernel<<<1, 256, 0, stream>>>(stats, 128);
        bn_stats_kernel<<<512, 256, 0, stream>>>(hbuf, stats, N_NODES);
        bn_finalize_kernel<<<1, 64, 0, stream>>>(stats, gm[l], bt[l], ss);
        bn_apply_pool_kernel<<<NB_FEAT, 256, 0, stream>>>(hbuf, ss, xout[l], batch, pool, l * DIM);
    }

    pool_final_kernel<<<NB_POOL, 256, 0, stream>>>(pool, cnt, out);
    (void)in_sizes; (void)n_in; (void)out_size; (void)ws_size;
}
